// SVC8_5x5_70377334112434
// MI455X (gfx1250) — matrix operand order repaired and checked by reading, not compiled, not run on hardware
//
#include <hip/hip_runtime.h>

typedef __attribute__((ext_vector_type(16))) _Float16 v16h;
typedef __attribute__((ext_vector_type(8)))  float    v8f;

#define CIN   3
#define HH    720
#define WW    1280
#define COUT  64
#define KREAL 75          // Cin*5*5
#define XPITCH 136        // LDS input tile pitch (halfs), 132 used
#define OPITCH 132        // LDS output tile pitch (floats), 128 used

// ---------------------------------------------------------------------------
// Pack weights [8,64,3,5,5] f32 -> f16 WMMA A-fragments in ws.
// Fragment layout (16-bit A 16x32, wave32): lane%16 = M, element e of v16h
// holds K = kt*32 + 8*(lane/16) + e + (e>=8 ? 8 : 0).  K >= 75 zero-padded.
// ws layout: [class(8)][kt(3)][mt(4)][lane(32)][16 halfs]  = 96 KB total.
// ---------------------------------------------------------------------------
__global__ __launch_bounds__(256) void svc8_prep_weights(
    const float* __restrict__ w, _Float16* __restrict__ wf)
{
  const int f    = blockIdx.x * 8 + (threadIdx.x >> 5);   // 96 fragments
  const int lane = threadIdx.x & 31;
  if (f >= 96) return;
  const int cls  = f / 12;
  const int kt   = (f % 12) / 4;
  const int mt   = f % 4;
  const int half = lane >> 4;
  const int m    = lane & 15;
  const int cout = mt * 16 + m;

  _Float16* dst = wf + ((((cls * 3 + kt) * 4 + mt) * 32 + lane) * 16);
#pragma unroll
  for (int e = 0; e < 16; ++e) {
    int k = kt * 32 + 8 * half + e + ((e >= 8) ? 8 : 0);
    float v = 0.f;
    if (k < KREAL) {
      int ci = k / 25, rr = k % 25, kh = rr / 5, kw = rr % 5;
      v = w[(((cls * COUT + cout) * CIN + ci) * 5 + kh) * 5 + kw];
    }
    dst[e] = (_Float16)v;
  }
}

// ---------------------------------------------------------------------------
// Main spatially-varying conv.  Block = 8 waves; handles (b, row-class r,
// 4 rows h = r+4*(rg*4+it), 128 output cols).  Wave t: parity p=t&1 ->
// class 2r+p, 16 positions.  12 x v_wmma_f32_16x16x32_f16 per wave per row.
// ---------------------------------------------------------------------------
__global__ __launch_bounds__(256) void svc8_conv(
    const float* __restrict__ x, const _Float16* __restrict__ wf,
    const float* __restrict__ bias, float* __restrict__ out)
{
  __shared__ _Float16 xs[CIN * 5 * XPITCH];   // ~4.1 KB input tile (f16)
  __shared__ float    os[COUT * OPITCH];      // ~33.8 KB output tile (f32)

  const int tid  = threadIdx.x;
  const int wave = tid >> 5;
  const int lane = tid & 31;
  const int half = lane >> 4;
  const int n    = lane & 15;

  int bid = blockIdx.x;
  const int g  = bid % 10; bid /= 10;   // 128-wide column group
  const int rg = bid % 45; bid /= 45;   // row group (4 rows each)
  const int r  = bid % 4;  bid /= 4;    // row class (h % 4)
  const int b  = bid;                   // batch

  const int p   = wave & 1;             // column parity (w % 2)
  const int jt  = wave >> 1;            // 16-position tile within parity
  const int cls = 2 * r + p;
  const int w0  = g * 128;

  // A-fragments for this wave's class: 4 M-tiles x 3 K-steps (96 VGPRs).
  v16h A[4][3];
#pragma unroll
  for (int mt = 0; mt < 4; ++mt)
#pragma unroll
    for (int kt = 0; kt < 3; ++kt)
      A[mt][kt] = *(const v16h*)(wf + ((((cls * 3 + kt) * 4 + mt) * 32 + lane) * 16));

  // Bias per (mt, acc element): C/D layout has M = v + 8*half, N = lane%16.
  float bb[4][8];
#pragma unroll
  for (int mt = 0; mt < 4; ++mt)
#pragma unroll
    for (int v = 0; v < 8; ++v)
      bb[mt][v] = bias[cls * COUT + mt * 16 + v + 8 * half];

  const int cbase = 2 * (jt * 16 + n) + p;   // LDS column of this lane's pos

  for (int it = 0; it < 4; ++it) {
    const int h = r + 4 * (rg * 4 + it);

    __syncthreads();
    // Cooperative stage: x[b][ci][h-2..h+2][w0-2..w0+129] -> f16 LDS (zero pad)
    for (int idx = tid; idx < CIN * 5 * 132; idx += 256) {
      int col = idx % 132;
      int t2  = idx / 132;
      int dy  = t2 % 5;
      int ci  = t2 / 5;
      int yy  = h - 2 + dy;
      int xx  = w0 - 2 + col;
      float v = 0.f;
      if ((unsigned)yy < (unsigned)HH && (unsigned)xx < (unsigned)WW)
        v = x[((b * CIN + ci) * HH + yy) * WW + xx];
      xs[(ci * 5 + dy) * XPITCH + col] = (_Float16)v;
    }
    __syncthreads();

    // B-fragments (32x16, K x N): lane%16 = N, element e -> K = 8*half + e + (e>=8 ? 8 : 0).
    // LDS offsets are compile-time constants; only the half-select is runtime.
    v16h Bf[3];
#pragma unroll
    for (int kt = 0; kt < 3; ++kt) {
      v16h bf;
#pragma unroll
      for (int e = 0; e < 16; ++e) {
        const int k0 = kt * 32 + e + ((e >= 8) ? 8 : 0);        // half = 0
        const int k1 = k0 + 8;           // half = 1
        const int off0 = (k0 < KREAL)
            ? (((k0 / 25) * 5 + (k0 % 25) / 5) * XPITCH + (k0 % 25) % 5) : -1;
        const int off1 = (k1 < KREAL)
            ? (((k1 / 25) * 5 + (k1 % 25) / 5) * XPITCH + (k1 % 25) % 5) : -1;
        _Float16 v = (_Float16)0.f;
        if (off0 >= 0 && off1 >= 0) {
          v = xs[cbase + (half ? off1 : off0)];
        } else if (off0 >= 0) {            // upper half K is pad -> zero
          _Float16 t = xs[cbase + off0];
          v = half ? (_Float16)0.f : t;
        }                                  // both pad -> zero, no load
        bf[e] = v;
      }
      Bf[kt] = bf;
    }

    // 12 WMMAs: 64 Cout x 16 positions, K = 96 (75 real + pad)
    v8f acc[4];
#pragma unroll
    for (int mt = 0; mt < 4; ++mt) {
      v8f c = {};
#pragma unroll
      for (int kt = 0; kt < 3; ++kt)
        c = __builtin_amdgcn_wmma_f32_16x16x32_f16(false, A[mt][kt],
                                                   false, Bf[kt],
                                                   (short)0, c, false, false);
      acc[mt] = c;
    }

    // Stage results (+bias) into interleaved LDS tile (conflict-free pattern)
#pragma unroll
    for (int mt = 0; mt < 4; ++mt)
#pragma unroll
      for (int v = 0; v < 8; ++v)
        os[(mt * 16 + v + 8 * half) * OPITCH + cbase] = acc[mt][v] + bb[mt][v];

    __syncthreads();

    // Fully-coalesced writeback: out[b][cout][h][w0..w0+127] as float4
    for (int idx = tid; idx < COUT * 32; idx += 256) {
      int row = idx >> 5;
      int c4  = idx & 31;
      float4 val = *(const float4*)&os[row * OPITCH + c4 * 4];
      *(float4*)&out[((b * COUT + row) * HH + h) * WW + w0 + c4 * 4] = val;
    }
  }
}

extern "C" void kernel_launch(void* const* d_in, const int* in_sizes, int n_in,
                              void* d_out, int out_size, void* d_ws, size_t ws_size,
                              hipStream_t stream) {
  const float* x    = (const float*)d_in[0];   // [2,3,720,1280]
  const float* w    = (const float*)d_in[1];   // [8,64,3,5,5]
  const float* bias = (const float*)d_in[2];   // [8,64]
  float* out        = (float*)d_out;           // [2,64,720,1280]
  _Float16* wf      = (_Float16*)d_ws;         // 96 KB packed A-fragments

  svc8_prep_weights<<<12, 256, 0, stream>>>(w, wf);
  // grid = B(2) * r(4) * rowgroups(45) * colgroups(10) = 3600 blocks
  svc8_conv<<<2 * 4 * 45 * 10, 256, 0, stream>>>(x, wf, bias, out);
}
